// MultiHeadVectorQuantizer_48713519071307
// MI455X (gfx1250) — matrix operand order repaired and checked by reading, not compiled, not run on hardware
//
#include <hip/hip_runtime.h>
#include <hip/hip_bf16.h>

typedef __attribute__((ext_vector_type(16))) __bf16 v16bf;
typedef __attribute__((ext_vector_type(8)))  __bf16 v8bf;
typedef __attribute__((ext_vector_type(4)))  __bf16 v4bf;
typedef __attribute__((ext_vector_type(8)))  float  v8f;

#define N_TOK   32768          // B*T
#define D_MODEL 1024
#define HD      256            // head dim
#define NH      4
#define NCODE   64
#define TT      128            // tokens per block
#define RS      264            // padded LDS row stride in bf16 elems (256 + 8)
#define LOSSOFF (N_TOK * D_MODEL)   // 33554432
#define IDXOFF  (LOSSOFF + 1)

__global__ __launch_bounds__(256)
void vq_main(const float* __restrict__ x,
             const float* __restrict__ emb,
             float* __restrict__ out,
             float* __restrict__ partials)
{
    __shared__ __bf16 xlds[TT * RS];       //  67.5 KB : token tile, bf16
    __shared__ __bf16 elds[NCODE * RS];    //  33.8 KB : codebook, bf16
    __shared__ float  insq[TT];
    __shared__ float  csq[NCODE];
    __shared__ int    sidx[TT];
    __shared__ float  lossacc;

    const int tid   = threadIdx.x;
    const int h     = blockIdx.y;
    const int tbase = blockIdx.x * TT;

    if (tid < TT)    insq[tid] = 0.f;
    if (tid < NCODE) csq[tid]  = 0.f;
    if (tid == 0)    lossacc   = 0.f;
    __syncthreads();

    // ---- stage codebook (fp32 -> bf16 LDS) + ||e||^2 ----
    const float4* embv = (const float4*)emb;
    for (int i = tid; i < NCODE * (HD / 4); i += 256) {
        int code = i >> 6, chunk = i & 63;
        float4 f = embv[(h * NCODE + code) * (HD / 4) + chunk];
        v4bf b; b[0] = (__bf16)f.x; b[1] = (__bf16)f.y;
                b[2] = (__bf16)f.z; b[3] = (__bf16)f.w;
        *(v4bf*)&elds[code * RS + chunk * 4] = b;
        atomicAdd(&csq[code], f.x*f.x + f.y*f.y + f.z*f.z + f.w*f.w);
    }

    // ---- stage input tile (fp32 -> bf16 LDS) + ||x||^2 ----
    const float4* xv = (const float4*)x;
    for (int i = tid; i < TT * (HD / 4); i += 256) {
        int tl = i >> 6, chunk = i & 63;
        float4 f = xv[(size_t)(tbase + tl) * (D_MODEL / 4) + h * (HD / 4) + chunk];
        v4bf b; b[0] = (__bf16)f.x; b[1] = (__bf16)f.y;
                b[2] = (__bf16)f.z; b[3] = (__bf16)f.w;
        *(v4bf*)&xlds[tl * RS + chunk * 4] = b;
        atomicAdd(&insq[tl], f.x*f.x + f.y*f.y + f.z*f.z + f.w*f.w);
    }
    __syncthreads();

    // ---- WMMA: product[16 tok x 64 codes], K = 256 in 8 steps of 32 ----
    const int wave = tid >> 5;
    const int lane = tid & 31;
    const int col  = lane & 15;     // N index within tile / A row within tile
    const int hi   = lane >> 4;

    v8f acc[4] = {};                          // 4 N-tiles of 16 codes
    const int arow = wave * 16 + col;         // A-matrix row (token)

#pragma unroll
    for (int kk = 0; kk < 8; ++kk) {
        // A fragment: lane<16 holds K {0..7,16..23}; lane>=16 holds {8..15,24..31}
        const int ac = kk * 32 + hi * 8;
        v8bf alo = *(const v8bf*)&xlds[arow * RS + ac];
        v8bf ahi = *(const v8bf*)&xlds[arow * RS + ac + 16];
        v16bf a = __builtin_shufflevector(alo, ahi,
                    0,1,2,3,4,5,6,7,8,9,10,11,12,13,14,15);
        // B fragment: column n = code, same K order as A ({0..7,16..23} / {8..15,24..31})
        const int bc = kk * 32 + hi * 8;
#pragma unroll
        for (int nt = 0; nt < 4; ++nt) {
            const int code = nt * 16 + col;
            v8bf blo = *(const v8bf*)&elds[code * RS + bc];
            v8bf bhi = *(const v8bf*)&elds[code * RS + bc + 16];
            v16bf b = __builtin_shufflevector(blo, bhi,
                        0,1,2,3,4,5,6,7,8,9,10,11,12,13,14,15);
            acc[nt] = __builtin_amdgcn_wmma_f32_16x16x32_bf16(
                        false, a, false, b, (short)0, acc[nt], false, false);
        }
    }

    // ---- argmin over 64 codes; score = ||e||^2 - 2<x,e> (||x||^2 const in k) ----
    float cs[4];
#pragma unroll
    for (int nt = 0; nt < 4; ++nt) cs[nt] = csq[nt * 16 + col];

    float lossPart = 0.f;
#pragma unroll
    for (int r = 0; r < 8; ++r) {
        float bv = cs[0] - 2.f * acc[0][r];
        int   bi = col;
#pragma unroll
        for (int nt = 1; nt < 4; ++nt) {
            float v = cs[nt] - 2.f * acc[nt][r];
            if (v < bv) { bv = v; bi = nt * 16 + col; }
        }
        // reduce across the 16 lanes of each half (C layout: lanes = N)
#pragma unroll
        for (int m = 1; m <= 8; m <<= 1) {
            float ov = __shfl_xor(bv, m, 32);
            int   oi = __shfl_xor(bi, m, 32);
            if (ov < bv || (ov == bv && oi < bi)) { bv = ov; bi = oi; }
        }
        if (col == 0) {                       // lanes 0 and 16
            int tl = wave * 16 + hi * 8 + r;  // C layout: M = r + 8*hi
            sidx[tl] = bi;
            lossPart += insq[tl] + bv;        // ||x-e||^2 at the argmin
        }
    }
    if (col == 0) atomicAdd(&lossacc, lossPart);
    __syncthreads();

    // ---- gather fp32 codebook rows -> quantized_st; write indices ----
    for (int i = tid; i < TT * (HD / 4); i += 256) {
        int tl = i >> 6, chunk = i & 63;
        int idx = sidx[tl];
        float4 q = embv[(h * NCODE + idx) * (HD / 4) + chunk];
        ((float4*)out)[(size_t)(tbase + tl) * (D_MODEL / 4) + h * (HD / 4) + chunk] = q;
        if (chunk == 0)
            out[(size_t)IDXOFF + (size_t)(tbase + tl) * NH + h] = (float)idx;
    }
    if (tid == 0)
        partials[blockIdx.y * gridDim.x + blockIdx.x] = lossacc;
}

__global__ __launch_bounds__(256)
void vq_loss_reduce(const float* __restrict__ partials, float* __restrict__ out_loss,
                    int n, float scale)
{
    __shared__ float s[256];
    float acc = 0.f;
    for (int i = threadIdx.x; i < n; i += 256) acc += partials[i];  // fixed order
    s[threadIdx.x] = acc;
    __syncthreads();
    for (int off = 128; off > 0; off >>= 1) {
        if ((int)threadIdx.x < off) s[threadIdx.x] += s[threadIdx.x + off];
        __syncthreads();
    }
    if (threadIdx.x == 0) *out_loss = s[0] * scale;
}

extern "C" void kernel_launch(void* const* d_in, const int* in_sizes, int n_in,
                              void* d_out, int out_size, void* d_ws, size_t ws_size,
                              hipStream_t stream)
{
    (void)in_sizes; (void)n_in; (void)out_size; (void)ws_size;
    const float* x   = (const float*)d_in[0];   // inputs (16,2048,1024) fp32
    const float* emb = (const float*)d_in[1];   // embeddings (4,64,256) fp32
    float* out       = (float*)d_out;
    float* partials  = (float*)d_ws;            // 1024 block partials

    dim3 grid(N_TOK / TT, NH);                  // 256 x 4 blocks
    vq_main<<<grid, 256, 0, stream>>>(x, emb, out, partials);

    const float scale = 1.5f / (float)(N_TOK * D_MODEL); // (1 + 0.5) * mean
    vq_loss_reduce<<<1, 256, 0, stream>>>(partials, out + LOSSOFF,
                                          NH * (N_TOK / TT), scale);
}
